// PerformerLayer_20521353740331
// MI455X (gfx1250) — compile-verified, matrix operand order checked by reading, not run on hardware
//
#include <hip/hip_runtime.h>

// ---------------------------------------------------------------------------
// Performer layer for MI455X (gfx1250, wave32, WMMA + TDM).
// Compute-bound on four 16384x1024x1024 GEMMs -> bf16 WMMA with fp32 accum.
// A/B tiles staged into LDS by the Tensor Data Mover (tensor_load_to_lds),
// double-buffered with s_wait_tensorcnt, so the GEMM inner loop is pure
// ds_load_b128 + v_wmma_f32_16x16x32_bf16.
// ---------------------------------------------------------------------------

#define D_MODEL 1024
#define HEADS   16
#define DEPTH   64
#define BATCH   4
#define SEQ     4096
#define ROWS    (BATCH * SEQ)      // 16384
#define EPS_K   1e-6f
#define EPS_LN  1e-6f

typedef __bf16 v16bf __attribute__((ext_vector_type(16)));
typedef float  v8f   __attribute__((ext_vector_type(8)));
typedef unsigned int v4u __attribute__((ext_vector_type(4)));
typedef int v8i __attribute__((ext_vector_type(8)));
typedef int v4i __attribute__((ext_vector_type(4)));

union Frag {
    v16bf v;
    unsigned int u[8];
    unsigned short h[16];
};

static __device__ inline v8f wmma_bf16(const v16bf& a, const v16bf& b, const v8f& c) {
    return __builtin_amdgcn_wmma_f32_16x16x32_bf16(false, a, false, b, (short)0, c, false, false);
}

static __device__ inline unsigned short f2bf(float f) {
    unsigned int x = __float_as_uint(f);
    unsigned int r = x + 0x7FFFu + ((x >> 16) & 1u);   // round-to-nearest-even
    return (unsigned short)(r >> 16);
}
static __device__ inline float bf2f(unsigned short h) {
    return __uint_as_float(((unsigned int)h) << 16);
}
static __device__ inline unsigned int pack2bf(float lo, float hi) {
    return (unsigned int)f2bf(lo) | ((unsigned int)f2bf(hi) << 16);
}

// Per-lane K offsets for the bf16 16x16x32 WMMA A/B fragment layout
// (ISA 7.12.2: lanes 0-15 hold K={0..7,16..23}, lanes 16-31 K={8..15,24..31}).
static __device__ inline int frag_kbase(int v, int hi) {
    return ((v < 4) ? (2 * v) : (2 * v + 8)) + hi * 8;
}

// ---------------------------------------------------------------------------
// TDM: async-load a 128-row x 32-col bf16 tile (row stride 1024 elements)
// from a 2D bf16 tensor into LDS at byte offset lds_off (contiguous, i.e.
// LDS image is [row][0..31] with stride 32 elements).
// D# layout per CDNA5 ISA ch.8: group0 {count,lds_addr,global_addr,type=2},
// group1 {data_size=2B, tensor_dim0/1, tile_dim0=32, tile_dim1=128, stride}.
// This toolchain exposes the 6-arg builtin (g0, g1, g2, g3, g4, cpol).
// ---------------------------------------------------------------------------
static __device__ inline void tdm_load_tile(const unsigned short* base,
                                            unsigned lds_off,
                                            unsigned row0, unsigned k0,
                                            unsigned rows_total) {
    const unsigned row_len = D_MODEL;   // elements per row
    unsigned long long ga = (unsigned long long)base +
                            ((unsigned long long)row0 * row_len + k0) * 2ull;
    v4u g0;
    g0[0] = 1u;                                        // count=1 valid
    g0[1] = lds_off;                                   // lds_addr (bytes)
    g0[2] = (unsigned)ga;                              // global_addr[31:0]
    g0[3] = (unsigned)((ga >> 32) & 0x1FFFFFFu)        // global_addr[56:32]
            | (2u << 30);                              // type=2 (image)
    v8i g1;
    g1[0] = (int)(1u << 16);                           // data_size=1 -> 2 bytes
    g1[1] = (int)((row_len & 0xFFFFu) << 16);          // tensor_dim0[15:0]
    g1[2] = (int)((row_len >> 16) |
                  ((rows_total & 0xFFFFu) << 16));     // dim0 hi / dim1 lo
    g1[3] = (int)((rows_total >> 16) | (32u << 16));   // dim1 hi / tile_dim0=32
    g1[4] = 128;                                       // tile_dim1=128, tile_dim2=0
    g1[5] = (int)row_len;                              // tensor_dim0_stride lo32
    g1[6] = 0;
    g1[7] = 0;
    v4i z4 = {0, 0, 0, 0};
    v8i z8 = {0, 0, 0, 0, 0, 0, 0, 0};
    __builtin_amdgcn_tensor_load_to_lds(g0, g1, z4, z4, z8, 0);
}

// ---------------------------------------------------------------------------
// Convert fp32 -> bf16, row-major (for q,k,v activations).
// ---------------------------------------------------------------------------
__global__ __launch_bounds__(256) void conv_bf16_kernel(const float* __restrict__ in,
                                                        unsigned short* __restrict__ out) {
    int idx = blockIdx.x * 256 + threadIdx.x;           // float4 index
    float4 f = *(const float4*)&in[(size_t)idx * 4];
    uint2 p;
    p.x = pack2bf(f.x, f.y);
    p.y = pack2bf(f.z, f.w);
    *(uint2*)&out[(size_t)idx * 4] = p;
}

// ---------------------------------------------------------------------------
// Transpose-convert 1024x1024 fp32 weight -> bf16 [n][k] (N x K).
// 32x32 LDS tiles, block (32,8).
// ---------------------------------------------------------------------------
__global__ __launch_bounds__(256) void transp_bf16_kernel(const float* __restrict__ in,
                                                          unsigned short* __restrict__ out) {
    __shared__ float t[32][33];
    const int k0 = blockIdx.x * 32;
    const int n0 = blockIdx.y * 32;
    const int tx = threadIdx.x, ty = threadIdx.y;
#pragma unroll
    for (int i = 0; i < 4; ++i)
        t[ty + i * 8][tx] = in[(k0 + ty + i * 8) * D_MODEL + n0 + tx];
    __syncthreads();
#pragma unroll
    for (int i = 0; i < 4; ++i)
        out[(n0 + ty + i * 8) * D_MODEL + k0 + tx] = f2bf(t[tx][ty + i * 8]);
}

// ---------------------------------------------------------------------------
// GEMM: C[16384x1024] = A[16384x1024](bf16) @ Bt[1024x1024]^T(bf16, stored NxK).
// Block tile 128x128, BK=32, 256 threads (8 waves), wave tile 64x32.
// TDM double-buffered LDS staging: wave0 pulls A tiles, wave1 pulls Bt tiles.
// Dynamic LDS: 4 x 8KB buffers (A0,B0,A1,B1) = 32KB.
// ---------------------------------------------------------------------------
__global__ __launch_bounds__(256) void gemm_tdm_kernel(const unsigned short* __restrict__ A,
                                                       const unsigned short* __restrict__ Bt,
                                                       float* __restrict__ C) {
    extern __shared__ __align__(16) char smem[];
    unsigned short* lds = (unsigned short*)smem;

    const int tid  = threadIdx.x;
    const int lane = tid & 31;
    const int wave = tid >> 5;
    const int bm = blockIdx.y * 128;
    const int bn = blockIdx.x * 128;
    const int wm = (wave & 1) * 64;
    const int wn = (wave >> 1) * 32;
    const int mIdx = lane & 15;
    const int hi   = lane >> 4;

    int kb[8];
#pragma unroll
    for (int v = 0; v < 8; ++v) kb[v] = frag_kbase(v, hi);

    v8f acc[4][2];
#pragma unroll
    for (int i = 0; i < 4; ++i)
#pragma unroll
        for (int j = 0; j < 2; ++j) acc[i][j] = (v8f){0,0,0,0,0,0,0,0};

    // prologue: fill buffer 0
    if (wave == 0)      tdm_load_tile(A,  0u,     (unsigned)bm, 0u, ROWS);
    else if (wave == 1) tdm_load_tile(Bt, 8192u,  (unsigned)bn, 0u, D_MODEL);
    __builtin_amdgcn_s_wait_tensorcnt(0);
    __syncthreads();

    const int NSTEP = D_MODEL / 32;     // 32
    for (int step = 0; step < NSTEP; ++step) {
        const int cur = step & 1;
        // kick off next tiles into the other buffer
        if (step + 1 < NSTEP) {
            unsigned nb = (unsigned)(cur ^ 1) * 16384u;   // byte offset
            unsigned kn = (unsigned)(step + 1) * 32u;
            if (wave == 0)      tdm_load_tile(A,  nb,         (unsigned)bm, kn, ROWS);
            else if (wave == 1) tdm_load_tile(Bt, nb + 8192u, (unsigned)bn, kn, D_MODEL);
        }

        const unsigned short* As = lds + cur * 8192;   // elements (16KB/buf pair)
        const unsigned short* Bs = As + 4096;          // +8KB

        Frag a[4], b[2];
#pragma unroll
        for (int i = 0; i < 4; ++i) {
            int row = wm + i * 16 + mIdx;
#pragma unroll
            for (int v = 0; v < 8; ++v)
                a[i].u[v] = *(const unsigned int*)&As[row * 32 + kb[v]];
        }
#pragma unroll
        for (int j = 0; j < 2; ++j) {
            int col = wn + j * 16 + mIdx;
#pragma unroll
            for (int v = 0; v < 8; ++v)
                b[j].u[v] = *(const unsigned int*)&Bs[col * 32 + kb[v]];
        }
#pragma unroll
        for (int i = 0; i < 4; ++i)
#pragma unroll
            for (int j = 0; j < 2; ++j)
                acc[i][j] = wmma_bf16(a[i].v, b[j].v, acc[i][j]);

        // wave0/1: next-buffer TDM done; then everyone syncs before reusing LDS
        __builtin_amdgcn_s_wait_tensorcnt(0);
        __syncthreads();
    }

    // epilogue: C layout lane0-15 M=r, lane16-31 M=8+r
#pragma unroll
    for (int i = 0; i < 4; ++i) {
        int m0 = bm + wm + i * 16 + hi * 8;
#pragma unroll
        for (int j = 0; j < 2; ++j) {
            int n = bn + wn + j * 16 + mIdx;
#pragma unroll
            for (int r = 0; r < 8; ++r) C[(m0 + r) * D_MODEL + n] = acc[i][j][r];
        }
    }
}

// ---------------------------------------------------------------------------
// LayerNorm + feature map. One block (256 threads) per token row.
// mode 0: q' = relu(ln)+eps         -> bf16 scattered to (b,h,n,d)
// mode 1: k' = (relu(ln)+eps)*mask  -> bf16 scattered to (b,h,n,d)
// mode 2: v  = ln*mask              -> bf16 scattered to (b,h,n,d)
// mode 3: ln(Y + bias)              -> fp32 to d_out (b,n,c)
// ---------------------------------------------------------------------------
__global__ __launch_bounds__(256) void ln_fm_kernel(const float* __restrict__ Y,
                                                    const float* __restrict__ gamma,
                                                    const float* __restrict__ beta,
                                                    const float* __restrict__ mask,
                                                    const float* __restrict__ bias,
                                                    int mode,
                                                    unsigned short* __restrict__ obf,
                                                    float* __restrict__ of32) {
    __shared__ float r1[256], r2[256];
    const int row = blockIdx.x;             // b*SEQ + n
    const int tid = threadIdx.x;

    float x[4];
    float s1 = 0.f, s2 = 0.f;
#pragma unroll
    for (int i = 0; i < 4; ++i) {
        int c = tid + i * 256;
        float v = Y[row * D_MODEL + c];
        if (mode == 3) v += bias[c];
        x[i] = v;
        s1 += v;
        s2 += v * v;
    }
    r1[tid] = s1; r2[tid] = s2;
    __syncthreads();
    for (int off = 128; off > 0; off >>= 1) {
        if (tid < off) { r1[tid] += r1[tid + off]; r2[tid] += r2[tid + off]; }
        __syncthreads();
    }
    const float mean = r1[0] * (1.0f / D_MODEL);
    const float var  = r2[0] * (1.0f / D_MODEL) - mean * mean;
    const float rstd = rsqrtf(var + EPS_LN);
    const float mk   = (mode == 1 || mode == 2) ? mask[row] : 1.0f;
    const int b = row >> 12;                 // row / SEQ
    const int n = row & (SEQ - 1);

#pragma unroll
    for (int i = 0; i < 4; ++i) {
        int c = tid + i * 256;
        float g = (x[i] - mean) * rstd * gamma[c] + beta[c];
        if (mode == 3) {
            of32[row * D_MODEL + c] = g;
        } else {
            float o;
            if (mode == 0)      o = fmaxf(g, 0.f) + EPS_K;
            else if (mode == 1) o = (fmaxf(g, 0.f) + EPS_K) * mk;
            else                o = g * mk;
            int h = c >> 6, d = c & 63;
            obf[(((b * HEADS + h) * SEQ) + n) * DEPTH + d] = f2bf(o);
        }
    }
}

// ---------------------------------------------------------------------------
// kv[bh][d][e] = sum_n k'[bh][n][d] * v[bh][n][e] ; ksum[bh][d] = sum_n k'.
// One block per (b,h); 128-row LDS tiles; thread t owns (d = t/4, 16 e's).
// ---------------------------------------------------------------------------
__global__ __launch_bounds__(256) void kv_ksum_kernel(const unsigned short* __restrict__ kp,
                                                      const unsigned short* __restrict__ vp,
                                                      float* __restrict__ kv,
                                                      float* __restrict__ ksum) {
    __shared__ __align__(16) unsigned short Ks[128 * DEPTH];
    __shared__ __align__(16) unsigned short Vs[128 * DEPTH];
    const int bh  = blockIdx.x;
    const int tid = threadIdx.x;
    const int d   = tid >> 2;
    const int e0  = (tid & 3) * 16;
    const unsigned short* kbase = kp + (size_t)bh * SEQ * DEPTH;
    const unsigned short* vbase = vp + (size_t)bh * SEQ * DEPTH;

    float s[16];
#pragma unroll
    for (int j = 0; j < 16; ++j) s[j] = 0.f;
    float ks = 0.f;

    for (int n0 = 0; n0 < SEQ; n0 += 128) {
#pragma unroll
        for (int i = 0; i < 4; ++i) {
            int idx = tid + i * 256;          // 1024 chunks of 8 bf16
            int nn = idx >> 3, c8 = (idx & 7) * 8;
            *(uint4*)&Ks[nn * DEPTH + c8] = *(const uint4*)&kbase[(n0 + nn) * DEPTH + c8];
            *(uint4*)&Vs[nn * DEPTH + c8] = *(const uint4*)&vbase[(n0 + nn) * DEPTH + c8];
        }
        __syncthreads();
        for (int nn = 0; nn < 128; ++nn) {
            float kval = bf2f(Ks[nn * DEPTH + d]);
            if ((tid & 3) == 0) ks += kval;
#pragma unroll
            for (int j = 0; j < 16; ++j)
                s[j] += kval * bf2f(Vs[nn * DEPTH + e0 + j]);
        }
        __syncthreads();
    }
#pragma unroll
    for (int j = 0; j < 16; ++j) kv[(bh * DEPTH + d) * DEPTH + e0 + j] = s[j];
    if ((tid & 3) == 0) ksum[bh * DEPTH + d] = ks;
}

// ---------------------------------------------------------------------------
// z[bh][n] = 1 / (sum_d q'[bh][n][d] * ksum[bh][d] + eps)
// ---------------------------------------------------------------------------
__global__ __launch_bounds__(256) void z_kernel(const unsigned short* __restrict__ qp,
                                                const float* __restrict__ ksum,
                                                float* __restrict__ z) {
    int flat = blockIdx.x * 256 + threadIdx.x;   // bh*SEQ + n
    int bh = flat >> 12;
    float acc = EPS_K;
#pragma unroll 8
    for (int d = 0; d < DEPTH; ++d)
        acc += bf2f(qp[(size_t)flat * DEPTH + d]) * ksum[bh * DEPTH + d];
    z[flat] = 1.0f / acc;
}

// ---------------------------------------------------------------------------
// att[b][n][h*64+e] = z[bh][n] * sum_d q'[bh][n][d] * kv[bh][d][e]   (WMMA)
// grid (S/128, B*H); 8 waves x 16 rows each; kv staged transposed in LDS.
// ---------------------------------------------------------------------------
__global__ __launch_bounds__(256) void att_kernel(const unsigned short* __restrict__ qp,
                                                  const float* __restrict__ kv,
                                                  const float* __restrict__ z,
                                                  unsigned short* __restrict__ att) {
    __shared__ __align__(16) unsigned short KVs[DEPTH * DEPTH];  // [e][d]
    __shared__ float Zs[128];
    const int bh  = blockIdx.y;
    const int nb  = blockIdx.x * 128;
    const int tid = threadIdx.x;
    const int lane = tid & 31, wave = tid >> 5;
    const int mIdx = lane & 15, hi = lane >> 4;
    const int b = bh >> 4, h = bh & 15;

#pragma unroll
    for (int i = 0; i < 16; ++i) {
        int idx = tid + i * 256;              // 4096 elements
        int d = idx >> 6, e = idx & 63;
        KVs[e * DEPTH + d] = f2bf(kv[(bh * DEPTH + d) * DEPTH + e]);
    }
    if (tid < 128) Zs[tid] = z[bh * SEQ + nb + tid];
    __syncthreads();

    int kb[8];
#pragma unroll
    for (int v = 0; v < 8; ++v) kb[v] = frag_kbase(v, hi);

    const int r0 = nb + wave * 16;            // 16 rows per wave
    Frag a[2], bfr[2][4];
#pragma unroll
    for (int s = 0; s < 2; ++s) {
        int row = r0 + mIdx;
        const unsigned short* src = qp + ((size_t)bh * SEQ + row) * DEPTH + s * 32;
#pragma unroll
        for (int v = 0; v < 8; ++v) a[s].u[v] = *(const unsigned int*)&src[kb[v]];
#pragma unroll
        for (int j = 0; j < 4; ++j) {
            int e = j * 16 + mIdx;
#pragma unroll
            for (int v = 0; v < 8; ++v)
                bfr[s][j].u[v] = *(const unsigned int*)&KVs[e * DEPTH + s * 32 + kb[v]];
        }
    }

    v8f acc[4];
#pragma unroll
    for (int j = 0; j < 4; ++j) acc[j] = (v8f){0,0,0,0,0,0,0,0};
#pragma unroll
    for (int s = 0; s < 2; ++s)
#pragma unroll
        for (int j = 0; j < 4; ++j)
            acc[j] = wmma_bf16(a[s].v, bfr[s][j].v, acc[j]);

#pragma unroll
    for (int j = 0; j < 4; ++j) {
        int e = j * 16 + mIdx;
#pragma unroll
        for (int r = 0; r < 8; ++r) {
            int lrow = wave * 16 + hi * 8 + r;
            int n = nb + lrow;
            float val = acc[j][r] * Zs[lrow];
            att[((size_t)(b * SEQ + n)) * D_MODEL + h * DEPTH + e] = f2bf(val);
        }
    }
}

// ---------------------------------------------------------------------------
extern "C" void kernel_launch(void* const* d_in, const int* in_sizes, int n_in,
                              void* d_out, int out_size, void* d_ws, size_t ws_size,
                              hipStream_t stream) {
    const float* q       = (const float*)d_in[0];
    const float* k       = (const float*)d_in[1];
    const float* v       = (const float*)d_in[2];
    const float* mask    = (const float*)d_in[3];
    const float* wq      = (const float*)d_in[4];
    const float* wk      = (const float*)d_in[5];
    const float* wv      = (const float*)d_in[6];
    const float* gamma   = (const float*)d_in[7];
    const float* beta    = (const float*)d_in[8];
    const float* dense_w = (const float*)d_in[9];
    const float* dense_b = (const float*)d_in[10];
    float* out = (float*)d_out;

    // workspace carve-up (~300 MB)
    char* ws = (char*)d_ws;
    unsigned short* qb   = (unsigned short*)ws;  ws += (size_t)ROWS * D_MODEL * 2;     // 32MB
    unsigned short* kbuf = (unsigned short*)ws;  ws += (size_t)ROWS * D_MODEL * 2;     // 32MB
    unsigned short* vb   = (unsigned short*)ws;  ws += (size_t)ROWS * D_MODEL * 2;     // 32MB
    unsigned short* wqb  = (unsigned short*)ws;  ws += (size_t)D_MODEL * D_MODEL * 2;  // 2MB (transposed)
    unsigned short* wkb  = (unsigned short*)ws;  ws += (size_t)D_MODEL * D_MODEL * 2;
    unsigned short* wvb  = (unsigned short*)ws;  ws += (size_t)D_MODEL * D_MODEL * 2;
    unsigned short* dwb  = (unsigned short*)ws;  ws += (size_t)D_MODEL * D_MODEL * 2;
    float*          Y    = (float*)ws;           ws += (size_t)ROWS * D_MODEL * 4;     // 64MB
    unsigned short* qp   = (unsigned short*)ws;  ws += (size_t)ROWS * D_MODEL * 2;     // 32MB
    unsigned short* kp   = (unsigned short*)ws;  ws += (size_t)ROWS * D_MODEL * 2;     // 32MB
    unsigned short* vp   = (unsigned short*)ws;  ws += (size_t)ROWS * D_MODEL * 2;     // 32MB
    unsigned short* attb = (unsigned short*)ws;  ws += (size_t)ROWS * D_MODEL * 2;     // 32MB
    float*          kvb  = (float*)ws;           ws += (size_t)BATCH * HEADS * DEPTH * DEPTH * 4;
    float*          ksum = (float*)ws;           ws += (size_t)BATCH * HEADS * DEPTH * 4;
    float*          zb   = (float*)ws;           ws += (size_t)BATCH * HEADS * SEQ * 4;

    const int conv_grid = (ROWS * D_MODEL / 4) / 256;         // 16384
    const dim3 tr_grid(D_MODEL / 32, D_MODEL / 32);           // 32 x 32
    const dim3 tr_blk(32, 8);
    const dim3 gemm_grid(D_MODEL / 128, ROWS / 128);          // 8 x 128
    const dim3 att_grid(SEQ / 128, BATCH * HEADS);            // 32 x 64
    const size_t gemm_lds = 4 * 8192;                         // 32KB double-buffer

    // one-time bf16 conversions (activations row-major, weights transposed NxK)
    conv_bf16_kernel<<<conv_grid, 256, 0, stream>>>(q, qb);
    conv_bf16_kernel<<<conv_grid, 256, 0, stream>>>(k, kbuf);
    conv_bf16_kernel<<<conv_grid, 256, 0, stream>>>(v, vb);
    transp_bf16_kernel<<<tr_grid, tr_blk, 0, stream>>>(wq, wqb);
    transp_bf16_kernel<<<tr_grid, tr_blk, 0, stream>>>(wk, wkb);
    transp_bf16_kernel<<<tr_grid, tr_blk, 0, stream>>>(wv, wvb);
    transp_bf16_kernel<<<tr_grid, tr_blk, 0, stream>>>(dense_w, dwb);

    // q projection -> LN -> q'
    gemm_tdm_kernel<<<gemm_grid, 256, gemm_lds, stream>>>(qb, wqb, Y);
    ln_fm_kernel<<<ROWS, 256, 0, stream>>>(Y, gamma, beta, mask, dense_b, 0, qp, nullptr);
    // k projection -> LN -> k'
    gemm_tdm_kernel<<<gemm_grid, 256, gemm_lds, stream>>>(kbuf, wkb, Y);
    ln_fm_kernel<<<ROWS, 256, 0, stream>>>(Y, gamma, beta, mask, dense_b, 1, kp, nullptr);
    // v projection -> LN -> v*mask
    gemm_tdm_kernel<<<gemm_grid, 256, gemm_lds, stream>>>(vb, wvb, Y);
    ln_fm_kernel<<<ROWS, 256, 0, stream>>>(Y, gamma, beta, mask, dense_b, 2, vp, nullptr);
    // linear attention
    kv_ksum_kernel<<<BATCH * HEADS, 256, 0, stream>>>(kp, vp, kvb, ksum);
    z_kernel<<<(BATCH * HEADS * SEQ) / 256, 256, 0, stream>>>(qp, ksum, zb);
    att_kernel<<<att_grid, 256, 0, stream>>>(qp, kvb, zb, attb);
    // dense + final LN
    gemm_tdm_kernel<<<gemm_grid, 256, gemm_lds, stream>>>(attb, dwb, Y);
    ln_fm_kernel<<<ROWS, 256, 0, stream>>>(Y, gamma, beta, mask, dense_b, 3, nullptr, out);
}
